// RnnAutoencoder_48241072668824
// MI455X (gfx1250) — matrix operand order repaired and checked by reading, not compiled, not run on hardware
//
#include <hip/hip_runtime.h>
#include <hip/hip_bf16.h>

typedef __attribute__((ext_vector_type(16))) _Float16 v16h;
typedef __attribute__((ext_vector_type(8)))  _Float16 v8h;
typedef __attribute__((ext_vector_type(4)))  _Float16 v4h;
typedef __attribute__((ext_vector_type(8)))  float    v8f;

constexpr int Tc = 52;     // seq len
constexpr int Bc = 8192;   // batch
constexpr int WAVES = 2;   // waves per block (spread blocks over more WGPs)
constexpr int ROWS  = 16;  // batch rows per wave (WMMA M)

// ---- LDS layout ----
// block-shared weights (encoder phase, then overwritten with decoder):
//   wih  : [4][128][64] f16 = 65536 B,  whh : [4][128][32] f16 = 32768 B,  bias : [4][128] f32 = 2048 B
// per-wave scratch (8192 B): xbuf/ybuf [16][64] f16, y1 [16][64] f16, hb [4][16][32] f16
constexpr int    WIH_HALVES   = 4 * 128 * 64;
constexpr int    WHH_HALVES   = 4 * 128 * 32;
constexpr size_t LDS_WEIGHTS  = (size_t)WIH_HALVES * 2 + (size_t)WHH_HALVES * 2 + 4 * 128 * 4;
constexpr size_t LDS_PER_WAVE = (size_t)(1024 + 1024 + 2048) * 2;
constexpr size_t LDS_TOTAL    = LDS_WEIGHTS + WAVES * LDS_PER_WAVE;

// --- WMMA A-tile (16x32 f16) from LDS ---
static __device__ __forceinline__ v16h load_A(const _Float16* src, int stride, int kbase, int lane) {
  int row  = lane & 15;
  int koff = (lane & 16) ? 8 : 0;
  const _Float16* p = src + row * stride + kbase + koff;
  v8h lo = *(const v8h*)(p);
  v8h hi = *(const v8h*)(p + 16);
  return __builtin_shufflevector(lo, hi, 0, 1, 2, 3, 4, 5, 6, 7, 8, 9, 10, 11, 12, 13, 14, 15);
}

// --- WMMA B-tile (32x16 f16) = W^T tile: column n of B = row (ntile*16+n) of W, contiguous K ---
static __device__ __forceinline__ v16h load_B(const _Float16* w, int stride, int ntile, int kbase, int lane) {
  int col = lane & 15;
  int kh  = (lane & 16) ? 8 : 0;
  const _Float16* p = w + (ntile * 16 + col) * stride + kbase + kh;
  v8h lo = *(const v8h*)(p);
  v8h hi = *(const v8h*)(p + 16);
  return __builtin_shufflevector(lo, hi, 0, 1, 2, 3, 4, 5, 6, 7, 8, 9, 10, 11, 12, 13, 14, 15);
}

#if __has_builtin(__builtin_amdgcn_tanhf)
static __device__ __forceinline__ float fast_tanh(float v) { return __builtin_amdgcn_tanhf(v); }
#else
static __device__ __forceinline__ float fast_tanh(float v) {
  float e = __expf(2.0f * v);
  return 1.0f - 2.0f / (e + 1.0f);
}
#endif
// sigmoid(x) = 0.5*tanh(0.5*x) + 0.5  (1 trans op instead of exp+rcp)
static __device__ __forceinline__ float sigm_hb(float g, float halfbias) {
  return fmaf(0.5f, fast_tanh(fmaf(0.5f, g, halfbias)), 0.5f);
}

// One LSTM cell step for a 16-row batch tile.
// bv: 8 per-lane bias scalars (hoisted out of the time loop); acc starts at inline-0 C.
static __device__ __forceinline__ void lstm_cell(const _Float16* Wih, const _Float16* Whh,
                                                 const float* bv, const _Float16* xb,
                                                 const _Float16* hbf, v8f* c, v8f* h2, int lane) {
  v16h ax0 = load_A(xb, 64, 0, lane);
  v16h ax1 = load_A(xb, 64, 32, lane);
  v16h ah  = load_A(hbf, 32, 0, lane);
  v8f g[8];
#pragma unroll
  for (int n = 0; n < 8; n++) {
    v8f acc = {};  // folds to WMMA inline SRC2 = 0 (no broadcast movs)
    acc = __builtin_amdgcn_wmma_f32_16x16x32_f16(false, ax0, false, load_B(Wih, 64, n, 0, lane),
                                                 (short)0, acc, false, false);
    acc = __builtin_amdgcn_wmma_f32_16x16x32_f16(false, ax1, false, load_B(Wih, 64, n, 32, lane),
                                                 (short)0, acc, false, false);
    acc = __builtin_amdgcn_wmma_f32_16x16x32_f16(false, ah, false, load_B(Whh, 32, n, 0, lane),
                                                 (short)0, acc, false, false);
    g[n] = acc;
  }
#pragma unroll
  for (int p = 0; p < 2; p++) {
#pragma unroll
    for (int r = 0; r < 8; r++) {
      float iv = sigm_hb(g[p][r],     0.5f * bv[p]);
      float fv = sigm_hb(g[2 + p][r], 0.5f * bv[2 + p]);
      float gv = fast_tanh(g[4 + p][r] + bv[4 + p]);
      float ov = sigm_hb(g[6 + p][r], 0.5f * bv[6 + p]);
      float cn = fv * c[p][r] + iv * gv;
      c[p][r]  = cn;
      h2[p][r] = ov * fast_tanh(cn);
    }
  }
}

// C-layout f32 tiles -> LDS [16][32] f16 (A-source layout for next step)
static __device__ __forceinline__ void store_h32(_Float16* hbf, const v8f* h2, int lane) {
  int col = lane & 15, rb = (lane & 16) ? 8 : 0;
#pragma unroll
  for (int p = 0; p < 2; p++)
#pragma unroll
    for (int r = 0; r < 8; r++) hbf[(rb + r) * 32 + p * 16 + col] = (_Float16)h2[p][r];
}

// C-layout tiles -> columns [coff, coff+32) of LDS [16][64] f16
static __device__ __forceinline__ void store_cols64(_Float16* buf, int coff, const v8f* h2, int lane) {
  int col = lane & 15, rb = (lane & 16) ? 8 : 0;
#pragma unroll
  for (int p = 0; p < 2; p++)
#pragma unroll
    for (int r = 0; r < 8; r++) buf[(rb + r) * 64 + coff + p * 16 + col] = (_Float16)h2[p][r];
}

// stage [16][64] f32 global timestep slab -> LDS f16
static __device__ __forceinline__ void stage16x64(_Float16* dst, const float* src,
                                                  int base, int t, int lane) {
  for (int i = lane; i < 1024; i += 32) {
    int row = i >> 6, u = i & 63;
    dst[i] = (_Float16)src[((size_t)(base + row) * Tc + t) * 64 + u];
  }
}

// C-layout tiles -> global [B][T][64] f32 at column offset coff
static __device__ __forceinline__ void stageout(float* dst, int base, int t, int coff,
                                                const v8f* h2, int lane) {
  int col = lane & 15, rb = (lane & 16) ? 8 : 0;
#pragma unroll
  for (int p = 0; p < 2; p++)
#pragma unroll
    for (int r = 0; r < 8; r++)
      dst[((size_t)(base + rb + r) * Tc + t) * 64 + coff + p * 16 + col] = h2[p][r];
}

// vectorized f32 -> f16 weight preload (float4 loads, 8B LDS stores)
static __device__ __forceinline__ void load_weights(const float* Wih, const float* Whh, const float* b,
                                                    _Float16* wih, _Float16* whh, float* bias,
                                                    int tid, int nthr) {
  const float4* s1 = (const float4*)Wih;
  const float4* s2 = (const float4*)Whh;
  v4h* d1 = (v4h*)wih;
  v4h* d2 = (v4h*)whh;
  for (int i = tid; i < WIH_HALVES / 4; i += nthr) {
    float4 v = s1[i];
    v4h h = {(_Float16)v.x, (_Float16)v.y, (_Float16)v.z, (_Float16)v.w};
    d1[i] = h;
  }
  for (int i = tid; i < WHH_HALVES / 4; i += nthr) {
    float4 v = s2[i];
    v4h h = {(_Float16)v.x, (_Float16)v.y, (_Float16)v.z, (_Float16)v.w};
    d2[i] = h;
  }
  for (int i = tid; i < 512; i += nthr) bias[i] = b[i];
}

__global__ __launch_bounds__(WAVES * 32) void rnn_ae_kernel(
    const float* __restrict__ x, const float* __restrict__ h0, const float* __restrict__ c0,
    const float* __restrict__ eWih, const float* __restrict__ eWhh, const float* __restrict__ eb,
    const float* __restrict__ dWih, const float* __restrict__ dWhh, const float* __restrict__ db,
    float* __restrict__ out) {
  extern __shared__ __align__(16) char smem[];
  _Float16* wih  = (_Float16*)smem;                  // [4][128][64]
  _Float16* whh  = wih + WIH_HALVES;                 // [4][128][32]
  float*    bias = (float*)(whh + WHH_HALVES);       // [4][128]
  char* scratch  = (char*)(bias + 512);

  const int tid  = threadIdx.x;
  const int wave = tid >> 5;
  const int lane = tid & 31;
  _Float16* xbuf = (_Float16*)(scratch + (size_t)wave * LDS_PER_WAVE);  // [16][64]
  _Float16* y1   = xbuf + 1024;                                          // [16][64]
  _Float16* hb   = y1 + 1024;                                            // [4][16][32]

  const int base = (blockIdx.x * WAVES + wave) * ROWS;
  const int col  = lane & 15;
  const int rb   = (lane & 16) ? 8 : 0;
  float* xhat = out;                        // [B][T][64]; also encoder-l0 staging
  float* z    = out + (size_t)Bc * Tc * 64; // [B][256]

  // ---------------- phase 1: encoder (fwd+bwd scans fused per layer) ----------------
  load_weights(eWih, eWhh, eb, wih, whh, bias, tid, blockDim.x);
  __syncthreads();

  for (int l = 0; l < 2; l++) {
    const int kf = 2 * l, kb = 2 * l + 1;
    const _Float16* W1f = wih + kf * 8192;
    const _Float16* W2f = whh + kf * 4096;
    const _Float16* W1b = wih + kb * 8192;
    const _Float16* W2b = whh + kb * 4096;

    // per-lane bias scalars, hoisted out of the time loop
    float bvf[8], bvb[8];
#pragma unroll
    for (int n = 0; n < 8; n++) {
      bvf[n] = bias[kf * 128 + n * 16 + col];
      bvb[n] = bias[kb * 128 + n * 16 + col];
    }
    // init h (LDS f16) and c (C-layout regs) for both directions
    for (int i = lane; i < 512; i += 32) {
      int row = i >> 5, u = i & 31;
      hb[i]       = (_Float16)h0[(size_t)kf * Bc * 32 + (size_t)(base + row) * 32 + u];
      hb[512 + i] = (_Float16)h0[(size_t)kb * Bc * 32 + (size_t)(base + row) * 32 + u];
    }
    v8f cf[2], cb[2], h2f[2], h2b[2];
#pragma unroll
    for (int p = 0; p < 2; p++)
#pragma unroll
      for (int r = 0; r < 8; r++) {
        cf[p][r] = c0[(size_t)kf * Bc * 32 + (size_t)(base + rb + r) * 32 + p * 16 + col];
        cb[p][r] = c0[(size_t)kb * Bc * 32 + (size_t)(base + rb + r) * 32 + p * 16 + col];
      }

    const float* src = (l == 0) ? x : xhat;
#pragma unroll 1
    for (int s = 0; s < Tc; s++) {
      const int tf = s, tb = Tc - 1 - s;
      stage16x64(xbuf, src, base, tf, lane);  // fwd input
      stage16x64(y1, src, base, tb, lane);    // bwd input
      if (s + 1 < Tc) {  // warm next step's rows (global_prefetch_b8)
        int row = lane & 15;
        const float* pf = (lane < 16)
            ? &src[((size_t)(base + row) * Tc + (tf + 1)) * 64]
            : &src[((size_t)(base + row) * Tc + (tb - 1)) * 64];
        __builtin_prefetch(pf, 0, 1);
      }
      // two independent chains: scheduler interleaves WMMA / VALU / DS streams
      lstm_cell(W1f, W2f, bvf, xbuf, hb, cf, h2f, lane);
      lstm_cell(W1b, W2b, bvb, y1, hb + 512, cb, h2b, lane);
      store_h32(hb, h2f, lane);
      store_h32(hb + 512, h2b, lane);
      if (l == 0) {  // stage concat output for layer 1: fwd->cols 0-31 @tf, bwd->cols 32-63 @tb
        stageout(xhat, base, tf, 0, h2f, lane);
        stageout(xhat, base, tb, 32, h2b, lane);
      }
    }
    // final states -> z
#pragma unroll
    for (int p = 0; p < 2; p++)
#pragma unroll
      for (int r = 0; r < 8; r++) {
        size_t zrow = (size_t)(base + rb + r) * 256;
        z[zrow + kf * 32 + p * 16 + col]       = h2f[p][r];
        z[zrow + 128 + kf * 32 + p * 16 + col] = cf[p][r];
        z[zrow + kb * 32 + p * 16 + col]       = h2b[p][r];
        z[zrow + 128 + kb * 32 + p * 16 + col] = cb[p][r];
      }
  }

  // ---------------- phase 2: decoder ----------------
  __syncthreads();
  load_weights(dWih, dWhh, db, wih, whh, bias, tid, blockDim.x);
  __syncthreads();

  v8f cd[4][2], h2a[2], h2b2[2];
  float bv[4][8];
#pragma unroll
  for (int k = 0; k < 4; k++) {
#pragma unroll
    for (int n = 0; n < 8; n++) bv[k][n] = bias[k * 128 + n * 16 + col];
    for (int i = lane; i < 512; i += 32) {
      int row = i >> 5, u = i & 31;
      hb[k * 512 + i] = (_Float16)z[(size_t)(base + row) * 256 + k * 32 + u];
    }
#pragma unroll
    for (int p = 0; p < 2; p++)
#pragma unroll
      for (int r = 0; r < 8; r++)
        cd[k][p][r] = z[(size_t)(base + rb + r) * 256 + 128 + k * 32 + p * 16 + col];
  }
  _Float16* ybuf = xbuf;
  for (int i = lane; i < 1024; i += 32) ybuf[i] = (_Float16)0.0f;  // out0 = zeros

#pragma unroll 1
  for (int t = 0; t < Tc; t++) {
    // layer 0: both directions read ybuf, write y1 (independent chains)
    lstm_cell(wih + 0 * 8192, whh + 0 * 4096, bv[0], ybuf, hb + 0 * 512, cd[0], h2a, lane);
    lstm_cell(wih + 1 * 8192, whh + 1 * 4096, bv[1], ybuf, hb + 1 * 512, cd[1], h2b2, lane);
    store_h32(hb + 0 * 512, h2a, lane);
    store_h32(hb + 1 * 512, h2b2, lane);
    store_cols64(y1, 0, h2a, lane);
    store_cols64(y1, 32, h2b2, lane);
    // layer 1: read y1, write ybuf (next-step feedback) and x_hat
    lstm_cell(wih + 2 * 8192, whh + 2 * 4096, bv[2], y1, hb + 2 * 512, cd[2], h2a, lane);
    lstm_cell(wih + 3 * 8192, whh + 3 * 4096, bv[3], y1, hb + 3 * 512, cd[3], h2b2, lane);
    store_h32(hb + 2 * 512, h2a, lane);
    store_h32(hb + 3 * 512, h2b2, lane);
    store_cols64(ybuf, 0, h2a, lane);
    store_cols64(ybuf, 32, h2b2, lane);
    stageout(xhat, base, t, 0, h2a, lane);
    stageout(xhat, base, t, 32, h2b2, lane);
  }
}

extern "C" void kernel_launch(void* const* d_in, const int* in_sizes, int n_in,
                              void* d_out, int out_size, void* d_ws, size_t ws_size,
                              hipStream_t stream) {
  (void)in_sizes; (void)n_in; (void)out_size; (void)d_ws; (void)ws_size;
  const float* x    = (const float*)d_in[0];
  const float* h0   = (const float*)d_in[1];
  const float* c0   = (const float*)d_in[2];
  const float* eWih = (const float*)d_in[3];
  const float* eWhh = (const float*)d_in[4];
  const float* eb   = (const float*)d_in[5];
  const float* dWih = (const float*)d_in[6];
  const float* dWhh = (const float*)d_in[7];
  const float* db   = (const float*)d_in[8];
  float* out = (float*)d_out;

  static_assert(LDS_TOTAL <= 320 * 1024, "LDS budget");
  hipFuncSetAttribute((const void*)rnn_ae_kernel,
                      hipFuncAttributeMaxDynamicSharedMemorySize, (int)LDS_TOTAL);
  dim3 grid(Bc / (WAVES * ROWS));  // 256 blocks -> more WGPs busy
  dim3 block(WAVES * 32);          // 2 waves
  rnn_ae_kernel<<<grid, block, LDS_TOTAL, stream>>>(x, h0, c0, eWih, eWhh, eb, dWih, dWhh, db, out);
}
